// HydraMoRBlock_59657095741548
// MI455X (gfx1250) — compile-verified, matrix operand order checked by reading, not run on hardware
//
#include <hip/hip_runtime.h>

#define B_    4
#define L_    2048
#define D_    2048
#define H_    5472
#define KSEL  1024          // L * capacity_ratio(0.5)
#define MT    16            // token tile (WMMA M)
#define HC    96            // H chunk = 6 WMMA col-tiles; 5472 = 57*96
#define NCHUNK (H_ / HC)    // 57
#define XS_STRIDE D_
#define ACT_STRIDE HC
#define NT_H  (H_ / 16)     // 342 n-tiles across H
#define NT_D  (D_ / 16)     // 128 n-tiles across D
#define TILE_ELEMS 512      // 32 lanes * 16 bf16 per packed 32x16 B-tile

typedef __attribute__((ext_vector_type(16))) __bf16 v16bf;
typedef __attribute__((ext_vector_type(8)))  float  v8f;

union V16 {
    v16bf v;
    unsigned short u[16];
    uint4 q[2];
};

__device__ __forceinline__ unsigned short f2bf(float f) {
    union { float f; unsigned int u; } c; c.f = f;
    unsigned int u = c.u;
    u += 0x7FFFu + ((u >> 16) & 1u);   // round-to-nearest-even
    return (unsigned short)(u >> 16);
}

__device__ __forceinline__ float silu_f(float x) {
    return x / (1.0f + __expf(-x));
}

// ---------------------------------------------------------------- zero output
__global__ void zero_out_kernel(float4* __restrict__ out, int n4) {
    int i = blockIdx.x * blockDim.x + threadIdx.x;
    const int stride = gridDim.x * blockDim.x;
    for (; i < n4; i += stride) out[i] = make_float4(0.f, 0.f, 0.f, 0.f);
}

// ---------------------------------------------------------------- weight pack
// f32 [K][N] row-major  ->  bf16 packed 32x16 WMMA B-tiles.
// Tile (kt,nt) occupies TILE_ELEMS contiguous bf16; within it lane l
// (l = n%16 + 16*half) holds its 16 K-elements contiguously, matching the
// ISA 7.12.2 16-bit operand VGPR layout:
//   krel 0-7 ->half0 e0-7 | 8-15 ->half1 e0-7 | 16-23 ->half0 e8-15 | 24-31 ->half1 e8-15
__global__ void pack_b_kernel(const float* __restrict__ w,
                              unsigned short* __restrict__ wp,
                              int K, int N) {
    const size_t total = (size_t)K * N;
    size_t i = (size_t)blockIdx.x * blockDim.x + threadIdx.x;
    const size_t stride = (size_t)gridDim.x * blockDim.x;
    const int ntiles = N >> 4;
    for (; i < total; i += stride) {
        const int k = (int)(i / N);
        const int n = (int)(i - (size_t)k * N);
        const int kt   = k >> 5;
        const int krel = k & 31;
        const int half = (krel >> 3) & 1;
        const int e    = (krel & 7) + ((krel & 16) ? 8 : 0);
        const int lane = (n & 15) + (half << 4);
        const size_t dst = ((size_t)kt * ntiles + (n >> 4)) * TILE_ELEMS + lane * 16 + e;
        wp[dst] = f2bf(w[i]);
    }
}

// ---------------------------------------------------------------- router
__global__ void router_kernel(const float* __restrict__ x,
                              const float* __restrict__ wr,
                              const float* __restrict__ br,
                              float* __restrict__ scores) {
    const int wave = threadIdx.x >> 5;
    const int lane = threadIdx.x & 31;
    const int tok  = blockIdx.x * 8 + wave;      // [0, B*L)
    const float4* xr = (const float4*)(x + (size_t)tok * D_);
    const float4* w4 = (const float4*)wr;
    float s = 0.f;
    for (int j = lane; j < D_ / 4; j += 32) {
        float4 a = xr[j], b = w4[j];
        s += a.x * b.x + a.y * b.y + a.z * b.z + a.w * b.w;
    }
    for (int off = 16; off > 0; off >>= 1) s += __shfl_down(s, off, 32);
    if (lane == 0) scores[tok] = s + br[0];
}

// ---------------------------------------------------------------- top-k select
// Exact top-k=1024 per batch via rank counting with index tie-break
// (stable top_k), then compaction in ascending-index order (== sorted idx).
__global__ void select_kernel(const float* __restrict__ scores,
                              int* __restrict__ idx_sel) {
    __shared__ float ss[L_];
    __shared__ unsigned char sel[L_];
    __shared__ int psum[257];
    const int b   = blockIdx.x;
    const int tid = threadIdx.x;
    for (int i = tid; i < L_; i += 256) ss[i] = scores[b * L_ + i];
    __syncthreads();
    int cnt = 0;
    const int i0 = tid * 8;
    for (int t = 0; t < 8; ++t) {
        const int i = i0 + t;
        const float si = ss[i];
        int rank = 0;
        for (int j = 0; j < L_; ++j) {
            const float sj = ss[j];
            rank += (sj > si) || (sj == si && j < i);
        }
        const unsigned char f = (rank < KSEL) ? 1 : 0;
        sel[i] = f;
        cnt += f;
    }
    psum[tid + 1] = cnt;
    __syncthreads();
    if (tid == 0) {
        psum[0] = 0;
        for (int t = 0; t < 256; ++t) psum[t + 1] += psum[t];
    }
    __syncthreads();
    int pos = psum[tid];
    for (int t = 0; t < 8; ++t) {
        const int i = i0 + t;
        if (sel[i]) idx_sel[b * KSEL + pos++] = i;
    }
}

// ---------------------------------------------------------------- fused MoD MLP
// One workgroup (8 waves) per 16-token tile. bf16 WMMA, f32 accumulate.
// B operands stream straight from packed bf16 tiles: two b128 loads per tile.
__global__ void __launch_bounds__(256)
mod_mlp_kernel(const float* __restrict__ x,
               const unsigned short* __restrict__ wgp,   // packed [D/32][H/16] tiles
               const unsigned short* __restrict__ wup,
               const unsigned short* __restrict__ wdp,   // packed [H/32][D/16] tiles
               const int* __restrict__ idx_sel,
               float* __restrict__ out) {
    extern __shared__ char smem_raw[];
    unsigned short* xs  = (unsigned short*)smem_raw;          // [MT][D_]  bf16
    unsigned short* act = xs + MT * XS_STRIDE;                // [MT][HC]  bf16
    int* rowg = (int*)(act + MT * ACT_STRIDE);                // [MT]

    const int tile = blockIdx.x;          // 0..63  (KSEL/MT)
    const int b    = blockIdx.y;          // 0..3
    const int tid  = threadIdx.x;
    const int wave = tid >> 5;
    const int lane = tid & 31;
    const int half = lane >> 4;
    const int lm   = lane & 15;

    if (tid < MT) rowg[tid] = idx_sel[b * KSEL + tile * MT + tid];
    __syncthreads();

    // stage the 16 gathered x rows into LDS as bf16 (float4 in, uint2 out)
    for (int i = tid; i < MT * D_ / 4; i += 256) {
        const int r = i >> 9;             // / (D_/4)
        const int c = (i & (D_ / 4 - 1)) * 4;
        const float4 v = *(const float4*)(&x[((size_t)b * L_ + rowg[r]) * D_ + c]);
        uint2 p;
        p.x = (unsigned int)f2bf(v.x) | ((unsigned int)f2bf(v.y) << 16);
        p.y = (unsigned int)f2bf(v.z) | ((unsigned int)f2bf(v.w) << 16);
        *(uint2*)(&xs[r * XS_STRIDE + c]) = p;
    }
    __syncthreads();

    // persistent f32 accumulators: this wave owns 16 output col-tiles (256 cols)
    v8f acc[16];
#pragma unroll
    for (int j = 0; j < 16; ++j) acc[j] = v8f{};

    for (int chunk = 0; chunk < NCHUNK; ++chunk) {
        // prefetch next chunk's packed w_down tiles toward L2 (global_prefetch_b8)
        if (chunk + 1 < NCHUNK)
            __builtin_prefetch(&wdp[((size_t)(chunk + 1) * 3 * NT_D + wave * 16) * TILE_ELEMS
                                    + lane * 16], 0, 1);

        if (wave < 6) {
            // ---- phase 1: gate/up GEMM for this wave's 16 h-columns ----
            const int nt = chunk * 6 + wave;            // h-tile index in [0, NT_H)
            v8f ag = v8f{}, au = v8f{};
            for (int kk = 0; kk < D_ / 32; ++kk) {
                V16 a, bgt, but;
                a.q[0] = *(const uint4*)(&xs[lm * XS_STRIDE + kk * 32 + half * 8]);
                a.q[1] = *(const uint4*)(&xs[lm * XS_STRIDE + kk * 32 + 16 + half * 8]);
                const size_t bb = ((size_t)kk * NT_H + nt) * TILE_ELEMS + lane * 16;
                bgt.q[0] = *(const uint4*)(&wgp[bb]);
                bgt.q[1] = *(const uint4*)(&wgp[bb + 8]);
                but.q[0] = *(const uint4*)(&wup[bb]);
                but.q[1] = *(const uint4*)(&wup[bb + 8]);
                ag = __builtin_amdgcn_wmma_f32_16x16x32_bf16(
                        false, a.v, false, bgt.v, (short)0, ag, false, false);
                au = __builtin_amdgcn_wmma_f32_16x16x32_bf16(
                        false, a.v, false, but.v, (short)0, au, false, false);
            }
            // SwiGLU: act = silu(gate) * up  -> bf16 LDS tile [MT][HC]
#pragma unroll
            for (int i = 0; i < 8; ++i) {
                const int m = i + half * 8;
                act[m * ACT_STRIDE + wave * 16 + lm] = f2bf(silu_f(ag[i]) * au[i]);
            }
        }
        __syncthreads();

        // ---- phase 2: down-projection partial (K = 96) into persistent accs ----
#pragma unroll
        for (int kt = 0; kt < HC / 32; ++kt) {
            const int kb = kt * 32;
            const int ktg = chunk * 3 + kt;             // k-tile index in [0, H/32)
            V16 a;
            a.q[0] = *(const uint4*)(&act[lm * ACT_STRIDE + kb + half * 8]);
            a.q[1] = *(const uint4*)(&act[lm * ACT_STRIDE + kb + 16 + half * 8]);
#pragma unroll
            for (int j = 0; j < 16; ++j) {
                const size_t bb = ((size_t)ktg * NT_D + wave * 16 + j) * TILE_ELEMS + lane * 16;
                V16 bt;
                bt.q[0] = *(const uint4*)(&wdp[bb]);
                bt.q[1] = *(const uint4*)(&wdp[bb + 8]);
                acc[j] = __builtin_amdgcn_wmma_f32_16x16x32_bf16(
                            false, a.v, false, bt.v, (short)0, acc[j], false, false);
            }
        }
        __syncthreads();   // act is rewritten next chunk
    }

    // scatter f32 results to the gathered rows (C/D layout: lane=col, vgpr=row)
#pragma unroll
    for (int j = 0; j < 16; ++j) {
        const int d = (wave * 16 + j) * 16 + lm;
#pragma unroll
        for (int i = 0; i < 8; ++i) {
            const int m = i + half * 8;
            out[((size_t)b * L_ + rowg[m]) * D_ + d] = acc[j][i];
        }
    }
}

// ---------------------------------------------------------------- launcher
extern "C" void kernel_launch(void* const* d_in, const int* in_sizes, int n_in,
                              void* d_out, int out_size, void* d_ws, size_t ws_size,
                              hipStream_t stream) {
    const float* x  = (const float*)d_in[0];
    const float* wr = (const float*)d_in[1];
    const float* br = (const float*)d_in[2];
    const float* wg = (const float*)d_in[3];
    const float* wu = (const float*)d_in[4];
    const float* wd = (const float*)d_in[5];
    float* out = (float*)d_out;

    // workspace layout
    char* ws = (char*)d_ws;
    float* scores = (float*)ws;                                   //  32 KB
    ws += (size_t)B_ * L_ * sizeof(float);
    int* idx_sel = (int*)ws;                                      //  16 KB
    ws += (size_t)B_ * KSEL * sizeof(int);
    const size_t wbytes = (size_t)D_ * H_ * sizeof(unsigned short);  // 22.4 MB each
    unsigned short* wgp = (unsigned short*)ws; ws += wbytes;
    unsigned short* wup = (unsigned short*)ws; ws += wbytes;
    unsigned short* wdp = (unsigned short*)ws;

    // 1) zero output (non-selected tokens emit 0)
    zero_out_kernel<<<4096, 256, 0, stream>>>((float4*)out, (B_ * L_ * D_) / 4);

    // 2) pack weights: f32 -> bf16 WMMA-native 32x16 tiles (amortized, ~9 us HBM)
    pack_b_kernel<<<8192, 256, 0, stream>>>(wg, wgp, D_, H_);
    pack_b_kernel<<<8192, 256, 0, stream>>>(wu, wup, D_, H_);
    pack_b_kernel<<<8192, 256, 0, stream>>>(wd, wdp, H_, D_);

    // 3) router scores: one wave per token, 8 tokens per block
    router_kernel<<<(B_ * L_) / 8, 256, 0, stream>>>(x, wr, br, scores);

    // 4) exact top-k + ascending-index compaction, one block per batch
    select_kernel<<<B_, 256, 0, stream>>>(scores, idx_sel);

    // 5) fused SwiGLU MLP on selected tokens, bf16 WMMA from packed tiles
    const size_t smem = (size_t)(MT * XS_STRIDE + MT * ACT_STRIDE) * sizeof(unsigned short)
                      + (size_t)MT * sizeof(int);
    dim3 grid(KSEL / MT, B_);
    mod_mlp_kernel<<<grid, 256, smem, stream>>>(x, wgp, wup, wdp, idx_sel, out);
}
